// Transformer_83348135346331
// MI455X (gfx1250) — matrix operand order repaired and checked by reading, not compiled, not run on hardware
//
#include <hip/hip_runtime.h>

// ---------------------------------------------------------------------------
// CDNA5 (gfx1250) implementation of the token-pruning transformer.
// All GEMMs route through one bf16 WMMA kernel (v_wmma_f32_16x16x32_bf16),
// f32 accumulate, with fused epilogues (scale/bias/GELU/ReLU/sigmoid/residual).
// LDS tiles are pre-swizzled (A row-major, B transposed) so both fragments are
// contiguous per lane -> ds_load_b128; staging is double-buffered with a
// fast unpredicated path for fully in-bounds K-slabs.
// ---------------------------------------------------------------------------

typedef __attribute__((ext_vector_type(16))) __bf16    v16bf;
typedef __attribute__((ext_vector_type(8)))  float     v8f;
typedef __attribute__((ext_vector_type(8)))  unsigned  v8u;

#define FB_BIAS 1
#define FB_GELU 2
#define FB_RES  4
#define FB_TRB  8
#define FB_RELU 16
#define FB_SIG  32
#define FB_SCL  64

__device__ __forceinline__ unsigned bf16r(float f) {
  unsigned u = __builtin_bit_cast(unsigned, f);
  return (u + 0x7FFFu + ((u >> 16) & 1u)) >> 16;   // round-to-nearest-even
}
__device__ __forceinline__ unsigned pack2bf(float a, float b) {
  return (bf16r(a) & 0xFFFFu) | (bf16r(b) << 16);
}

// ---------------------------------------------------------------------------
// WMMA GEMM: C[M,N] = epilogue(A[M,K] * B[K,N]) ; optional B given as [N,K].
// Block = 128 threads (4 waves), tile 64x64, K-step 32, double-buffered LDS.
// Per-wave: 16 rows x 4 column tiles -> 4 v_wmma per K-step.
// LDS layout (packed 2xbf16 per dword, row stride 20 dwords):
//   As[buf][r][k/2]  : A tile rows (64 x 32)       -> A frag = 2x ds_load_b128
//   BsT[buf][n][k/2] : B tile TRANSPOSED (64 x 32) -> B frag = 2x ds_load_b128
// ---------------------------------------------------------------------------
__global__ __launch_bounds__(128)
void gemm_wmma_bf16(const float* __restrict__ A, const float* __restrict__ Bm,
                    const float* __restrict__ bias, const float* __restrict__ Rr,
                    float* __restrict__ C,
                    int M, int N, int K, int lda, int ldb, int ldc,
                    long long sA, long long sB, long long sC, long long sR,
                    int flags, float scale) {
  __shared__ unsigned As[2][64][20];
  __shared__ unsigned BsT[2][64][20];

  const int z = blockIdx.z;
  const float* Ab = A + (long long)z * sA;
  const float* Bb = Bm + (long long)z * sB;
  float*       Cb = C + (long long)z * sC;
  const float* Rb = (flags & FB_RES) ? (Rr + (long long)z * sR) : nullptr;

  const int bm = blockIdx.y * 64, bn = blockIdx.x * 64;
  const int tid = threadIdx.x;
  const int wid = tid >> 5, lane = tid & 31;
  const int m0 = lane & 15, hi = lane >> 4;
  const bool trb = (flags & FB_TRB) != 0;

  // staging assignments (same split for A and B): thread -> (row/col, 16-wide k slab)
  const int sr = tid >> 1, sc0 = (tid & 1) * 16;
  const int agr = bm + sr;  const bool arok = (agr < M);
  const int bgn = bn + sr;  const bool bnok = (bgn < N);

  auto stage = [&](int kt, int buf) {
    const bool fullK = (kt + 32 <= K);
    { // A tile
      const float* ap = Ab + (long long)agr * lda + kt + sc0;
      if (arok && fullK) {
#pragma unroll
        for (int j = 0; j < 8; ++j)
          As[buf][sr][(sc0 >> 1) + j] = pack2bf(ap[2 * j], ap[2 * j + 1]);
      } else {
#pragma unroll
        for (int j = 0; j < 8; ++j) {
          int gc = kt + sc0 + 2 * j;
          float v0 = (arok && gc < K)     ? ap[2 * j]     : 0.0f;
          float v1 = (arok && gc + 1 < K) ? ap[2 * j + 1] : 0.0f;
          As[buf][sr][(sc0 >> 1) + j] = pack2bf(v0, v1);
        }
      }
    }
    { // B tile (stored transposed: BsT[n][k])
      if (trb) {
        const float* bp = Bb + (long long)bgn * ldb + kt + sc0;
        if (bnok && fullK) {
#pragma unroll
          for (int j = 0; j < 8; ++j)
            BsT[buf][sr][(sc0 >> 1) + j] = pack2bf(bp[2 * j], bp[2 * j + 1]);
        } else {
#pragma unroll
          for (int j = 0; j < 8; ++j) {
            int gk = kt + sc0 + 2 * j;
            float v0 = (bnok && gk < K)     ? bp[2 * j]     : 0.0f;
            float v1 = (bnok && gk + 1 < K) ? bp[2 * j + 1] : 0.0f;
            BsT[buf][sr][(sc0 >> 1) + j] = pack2bf(v0, v1);
          }
        }
      } else {
        if (bnok && fullK) {
#pragma unroll
          for (int j = 0; j < 8; ++j) {
            int gk = kt + sc0 + 2 * j;
            BsT[buf][sr][(sc0 >> 1) + j] =
                pack2bf(Bb[(long long)gk * ldb + bgn], Bb[(long long)(gk + 1) * ldb + bgn]);
          }
        } else {
#pragma unroll
          for (int j = 0; j < 8; ++j) {
            int gk = kt + sc0 + 2 * j;
            float v0 = (bnok && gk < K)     ? Bb[(long long)gk * ldb + bgn]       : 0.0f;
            float v1 = (bnok && gk + 1 < K) ? Bb[(long long)(gk + 1) * ldb + bgn] : 0.0f;
            BsT[buf][sr][(sc0 >> 1) + j] = pack2bf(v0, v1);
          }
        }
      }
    }
  };

  v8f acc[4];
#pragma unroll
  for (int i = 0; i < 4; ++i)
#pragma unroll
    for (int e = 0; e < 8; ++e) acc[i][e] = 0.0f;

  stage(0, 0);
  __syncthreads();

  int buf = 0;
  for (int kt = 0; kt < K; kt += 32) {
    if (kt + 32 < K) stage(kt + 32, buf ^ 1);   // prefetch next slab into other buffer

    // A fragment (16x32, ISA 7.12.2): lane holds 8 bf16 at k=hi*8 and 8 at 16+hi*8
    v8u au;
    {
      const unsigned* ap = &As[buf][wid * 16 + m0][0];
#pragma unroll
      for (int q = 0; q < 4; ++q) au[q] = ap[hi * 4 + q];
#pragma unroll
      for (int q = 0; q < 4; ++q) au[4 + q] = ap[8 + hi * 4 + q];
    }
    v16bf af = __builtin_bit_cast(v16bf, au);

#pragma unroll
    for (int nt = 0; nt < 4; ++nt) {
      // B fragment (32x16): lane holds 8 bf16 at k=hi*8 and 8 at 16+hi*8 of column nt*16+m0
      v8u bu;
      {
        const unsigned* bp = &BsT[buf][nt * 16 + m0][0];
#pragma unroll
        for (int q = 0; q < 4; ++q) bu[q] = bp[hi * 4 + q];
#pragma unroll
        for (int q = 0; q < 4; ++q) bu[4 + q] = bp[8 + hi * 4 + q];
      }
      v16bf bfr = __builtin_bit_cast(v16bf, bu);
      acc[nt] = __builtin_amdgcn_wmma_f32_16x16x32_bf16(
          false, af, false, bfr, (short)0, acc[nt], false, false);
    }
    __syncthreads();
    buf ^= 1;
  }

#pragma unroll
  for (int nt = 0; nt < 4; ++nt) {
#pragma unroll
    for (int v = 0; v < 8; ++v) {
      int gr = bm + wid * 16 + v + 8 * hi;   // C layout: VGPR v -> row v+8*hi
      int gc = bn + nt * 16 + m0;
      if (gr < M && gc < N) {
        float val = acc[nt][v];
        if (flags & FB_SCL)  val *= scale;
        if (flags & FB_BIAS) val += bias[gc];
        if (flags & FB_GELU) val = 0.5f * val * (1.0f + erff(val * 0.70710678118f));
        if (flags & FB_RELU) val = fmaxf(val, 0.0f);
        if (flags & FB_SIG)  val = 1.0f / (1.0f + __expf(-val));
        if (flags & FB_RES)  val += Rb[(long long)gr * ldc + gc];
        Cb[(long long)gr * ldc + gc] = val;
      }
    }
  }
}

// ---------------------------------------------------------------------------
// LayerNorm over D=256 (one block / row, 256 threads)
// ---------------------------------------------------------------------------
__global__ void layernorm_k(const float* __restrict__ x, const float* __restrict__ g,
                            const float* __restrict__ b, float* __restrict__ y, int D) {
  __shared__ float red[256];
  long long row = blockIdx.x;
  int t = threadIdx.x;
  float v = x[row * D + t];
  red[t] = v; __syncthreads();
  for (int off = 128; off > 0; off >>= 1) { if (t < off) red[t] += red[t + off]; __syncthreads(); }
  float mu = red[0] / (float)D; __syncthreads();
  float xc = v - mu;
  red[t] = xc * xc; __syncthreads();
  for (int off = 128; off > 0; off >>= 1) { if (t < off) red[t] += red[t + off]; __syncthreads(); }
  float var = red[0] / (float)D;
  y[row * D + t] = xc * rsqrtf(var + 1e-5f) * g[t] + b[t];
}

// Softmax over contiguous rows (in-place)
__global__ void softmax_rows_k(float* __restrict__ x, int cols) {
  __shared__ float red[256];
  long long row = blockIdx.x;
  float* p = x + row * (long long)cols;
  int t = threadIdx.x;
  float m = -3.4e38f;
  for (int i = t; i < cols; i += 256) m = fmaxf(m, p[i]);
  red[t] = m; __syncthreads();
  for (int off = 128; off > 0; off >>= 1) { if (t < off) red[t] = fmaxf(red[t], red[t + off]); __syncthreads(); }
  m = red[0]; __syncthreads();
  float s = 0.0f;
  for (int i = t; i < cols; i += 256) { float e = __expf(p[i] - m); p[i] = e; s += e; }
  red[t] = s; __syncthreads();
  for (int off = 128; off > 0; off >>= 1) { if (t < off) red[t] += red[t + off]; __syncthreads(); }
  float inv = 1.0f / red[0];
  for (int i = t; i < cols; i += 256) p[i] *= inv;
}

// Strided row copy / add / gather utilities
__global__ void copy_rows_k(const float* __restrict__ src, float* __restrict__ dst, int D,
                            long long sRow, long long dRow, long long sBatch, long long dBatch) {
  int r = blockIdx.x, z = blockIdx.y;
  const float* sp = src + (long long)z * sBatch + (long long)r * sRow;
  float* dp = dst + (long long)z * dBatch + (long long)r * dRow;
  for (int t = threadIdx.x; t < D; t += blockDim.x) dp[t] = sp[t];
}

__global__ void add_rows_k(float* __restrict__ dst, const float* __restrict__ src, int D,
                           long long dRow, long long sRow, long long dBatch, long long sBatch) {
  int r = blockIdx.x, z = blockIdx.y;
  float* dp = dst + (long long)z * dBatch + (long long)r * dRow;
  const float* sp = src + (long long)z * sBatch + (long long)r * sRow;
  for (int t = threadIdx.x; t < D; t += blockDim.x) dp[t] += sp[t];
}

__global__ void gather_rows_k(const float* __restrict__ src, const int* __restrict__ idx,
                              float* __restrict__ dst, int D,
                              long long sBatch, long long dBatch, int idxStride) {
  int r = blockIdx.x, z = blockIdx.y;
  int j = idx[(long long)z * idxStride + r];
  const float* sp = src + (long long)z * sBatch + (long long)j * D;
  float* dp = dst + (long long)z * dBatch + (long long)r * D;
  for (int t = threadIdx.x; t < D; t += blockDim.x) dp[t] = sp[t];
}

// attns output: dots[:, :, :K, K:]  (dots is (B,H,n,n), pre-softmax, scaled)
__global__ void attn_slice_k(const float* __restrict__ dots, float* __restrict__ out,
                             int n, int Kt, int now) {
  long long r = blockIdx.x;                 // over B*H*Kt rows
  int i = (int)(r % Kt);
  long long bh = r / Kt;
  const float* src = dots + (bh * n + i) * (long long)n + Kt;
  float* dst = out + r * (long long)now;
  for (int c = threadIdx.x; c < now; c += 256) dst[c] = src[c];
}

// stage-1 pruning score: softmax-pool over H*K rows of dots[:, :, :K, K:]
__global__ void score1_k(const float* __restrict__ dots, float* __restrict__ score,
                         int n, int Kt, int now) {
  int b = blockIdx.y;
  int j = blockIdx.x * blockDim.x + threadIdx.x;
  if (j >= now) return;
  int R = 8 * Kt;
  float m = -3.4e38f;
  for (int r = 0; r < R; ++r) {
    int h = r / Kt, i = r % Kt;
    float v = dots[(((long long)b * 8 + h) * n + i) * n + Kt + j];
    m = fmaxf(m, v);
  }
  float se = 0.0f, ws = 0.0f;
  for (int r = 0; r < R; ++r) {
    int h = r / Kt, i = r % Kt;
    float v = dots[(((long long)b * 8 + h) * n + i) * n + Kt + j];
    float e = __expf(v - m);
    se += e; ws += v * e;
  }
  score[(long long)b * now + j] = ws / se;
}

// iterative block-argmax top-k (per batch row), descending, low-index tie-break
__global__ void topk_k(const float* __restrict__ score, int* __restrict__ out,
                       int now, int tgt) {
  __shared__ float s[1024];
  __shared__ float bv[256];
  __shared__ int   bi[256];
  int b = blockIdx.x, t = threadIdx.x;
  const float* sc = score + (long long)b * now;
  for (int i = t; i < now; i += 256) s[i] = sc[i];
  __syncthreads();
  for (int k = 0; k < tgt; ++k) {
    float best = -3.4e38f; int bidx = 0x7FFFFFFF;
    for (int i = t; i < now; i += 256)
      if (s[i] > best || (s[i] == best && i < bidx)) { best = s[i]; bidx = i; }
    bv[t] = best; bi[t] = bidx; __syncthreads();
    for (int off = 128; off > 0; off >>= 1) {
      if (t < off) {
        if (bv[t + off] > bv[t] || (bv[t + off] == bv[t] && bi[t + off] < bi[t])) {
          bv[t] = bv[t + off]; bi[t] = bi[t + off];
        }
      }
      __syncthreads();
    }
    if (t == 0) {
      int sel = (bi[0] >= now) ? 0 : bi[0];
      out[(long long)b * tgt + k] = sel;
      s[sel] = -3.4e38f;
    }
    __syncthreads();
  }
}

// idx==3: kp_tok = x[:, :1] + sparse_kp  -> 17 rows
__global__ void kp_insert_k(const float* __restrict__ x, const float* __restrict__ skp,
                            float* __restrict__ outx, long long xBatch, long long oBatch) {
  int b = blockIdx.y, j = blockIdx.x;
  const float* sp = x + (long long)b * xBatch;
  float* dp = outx + (long long)b * oBatch + (long long)j * 256;
  for (int t = threadIdx.x; t < 256; t += blockDim.x) dp[t] = sp[t] + skp[(long long)j * 256 + t];
}

// kp_tok -> (b,3,6,c) placed at the first 6 rows of each group of x2
__global__ void kp_to_x2_k(const float* __restrict__ x, float* __restrict__ x2,
                           long long xBatch, int n2) {
  int b = blockIdx.y, j = blockIdx.x;          // j in [0,18)
  int g = j / 6, i = j % 6;
  int src = (j == 0) ? 0 : (j - 1);
  const float* sp = x + (long long)b * xBatch + (long long)src * 256;
  float* dp = x2 + (((long long)(b * 3 + g)) * n2 + i) * 256;
  for (int t = threadIdx.x; t < 256; t += blockDim.x) dp[t] = sp[t];
}

// img = img_tok * cm ; posg = pos * cm   (cm: (b, now, 3*256))
__global__ void mul_gate_k(const float* __restrict__ imgtok, const float* __restrict__ postok,
                           const float* __restrict__ cm, float* __restrict__ img,
                           float* __restrict__ pg, int now, long long tokBatch) {
  int r = blockIdx.x, z = blockIdx.y;
  int b = z / 3, g = z % 3;
  const float* it = imgtok + (long long)b * tokBatch + (long long)r * 256;
  const float* pt = postok + ((long long)b * now + r) * 256;
  const float* cp = cm + ((long long)b * now + r) * 768 + g * 256;
  float* io = img + ((long long)z * now + r) * 256;
  float* po = pg + ((long long)z * now + r) * 256;
  for (int t = threadIdx.x; t < 256; t += blockDim.x) {
    float c = cp[t];
    io[t] = it[t] * c;
    po[t] = pt[t] * c;
  }
}

// pre-stage-2 score: pad row (-1e5) + last stage-1 attn slice, softmax-pool 48 rows
__global__ void score_pre2_k(const float* __restrict__ al, float* __restrict__ score, int now) {
  int z = blockIdx.y; int b = z / 3, g = z % 3;
  int j = blockIdx.x * blockDim.x + threadIdx.x;
  if (j >= now) return;
  float m = -3.4e38f;
  for (int h = 0; h < 8; ++h)
    for (int i = 0; i < 6; ++i) {
      int id2 = g * 6 + i;
      float v = (id2 == 0) ? -100000.0f
                           : al[(((long long)b * 8 + h) * 17 + (id2 - 1)) * now + j];
      m = fmaxf(m, v);
    }
  float se = 0.0f, ws = 0.0f;
  for (int h = 0; h < 8; ++h)
    for (int i = 0; i < 6; ++i) {
      int id2 = g * 6 + i;
      float v = (id2 == 0) ? -100000.0f
                           : al[(((long long)b * 8 + h) * 17 + (id2 - 1)) * now + j];
      float e = __expf(v - m);
      se += e; ws += v * e;
    }
  score[(long long)z * now + j] = ws / se;
}

// stage-2 group K/V assembly with broadcast kp keys (18 kp + n2-6 img keys)
__global__ void build_kv_k(const float* __restrict__ qkv, float* __restrict__ Kb,
                           float* __restrict__ Vb, int n2, int nk2) {
  int j = blockIdx.x, z = blockIdx.y;          // z = b*3+g
  int b = z / 3, g = z % 3;
  int sg, si;
  if (j < 18) { sg = j / 6; si = j % 6; }
  else        { sg = g;     si = 6 + (j - 18); }
  const float* src = qkv + (((long long)(b * 3 + sg)) * n2 + si) * 768;
  float* kd = Kb + ((long long)z * nk2 + j) * 256;
  float* vd = Vb + ((long long)z * nk2 + j) * 256;
  for (int t = threadIdx.x; t < 256; t += blockDim.x) {
    kd[t] = src[256 + t];
    vd[t] = src[512 + t];
  }
}

// mask: row (g==0,i==0) fully, and column j==0, set to -FLT_MAX
__global__ void group_mask_k(float* __restrict__ d2, int n2, int nk2, long long total) {
  long long idx = (long long)blockIdx.x * blockDim.x + threadIdx.x;
  if (idx >= total) return;
  int j = (int)(idx % nk2);
  long long r = idx / nk2;
  int i = (int)(r % n2);
  long long zh = r / n2;
  long long z = zh / 8;
  int g = (int)(z % 3);
  if (j == 0 || (g == 0 && i == 0)) d2[idx] = -3.402823466e38f;
}

// stage-2 pruning score: softmax-pool over (h,i<6) of dots2[..., 18:]
__global__ void score_group2_k(const float* __restrict__ d2, float* __restrict__ score,
                               int n2, int nk2) {
  int z = blockIdx.y;
  int now = n2 - 6;
  int j = blockIdx.x * blockDim.x + threadIdx.x;
  if (j >= now) return;
  float m = -3.4e38f;
  for (int h = 0; h < 8; ++h)
    for (int i = 0; i < 6; ++i) {
      float v = d2[(((long long)z * 8 + h) * n2 + i) * nk2 + 18 + j];
      m = fmaxf(m, v);
    }
  float se = 0.0f, ws = 0.0f;
  for (int h = 0; h < 8; ++h)
    for (int i = 0; i < 6; ++i) {
      float v = d2[(((long long)z * 8 + h) * n2 + i) * nk2 + 18 + j];
      float e = __expf(v - m);
      se += e; ws += v * e;
    }
  score[(long long)z * now + j] = ws / se;
}

// kp_body = x2[:, :, :6].reshape(b,18,c)[:,1:]
__global__ void kp_out_k(const float* __restrict__ x2, float* __restrict__ out, int n2) {
  int b = blockIdx.y, j = blockIdx.x;  // j in [0,17)
  int flat = j + 1;
  int g = flat / 6, i = flat % 6;
  const float* src = x2 + (((long long)(b * 3 + g)) * n2 + i) * 256;
  float* dst = out + ((long long)b * 17 + j) * 256;
  for (int t = threadIdx.x; t < 256; t += blockDim.x) dst[t] = src[t];
}

// ---------------------------------------------------------------------------
// Host orchestration
// ---------------------------------------------------------------------------
static inline void swapf(float*& a, float*& b) { float* t = a; a = b; b = t; }

extern "C" void kernel_launch(void* const* d_in, const int* in_sizes, int n_in,
                              void* d_out, int out_size, void* d_ws, size_t ws_size,
                              hipStream_t stream) {
  (void)in_sizes; (void)n_in; (void)out_size; (void)ws_size;
  // jax tree-flatten (sorted dict keys) input order:
  // 0:x 1:pos 2:sparse_kp 3..6: group_mask {b1,b2,w1,w2}
  // 7.. : stage1 layers x11, then stage2 layers x11, each sorted:
  //   b1,b2,bout,ln1_b,ln1_g,ln2_b,ln2_g,w1,w2,wout,wqkv
  const float* x_in  = (const float*)d_in[0];
  const float* pos_in = (const float*)d_in[1];
  const float* skp   = (const float*)d_in[2];
  const float* gm_b1 = (const float*)d_in[3];
  const float* gm_b2 = (const float*)d_in[4];
  const float* gm_w1 = (const float*)d_in[5];
  const float* gm_w2 = (const float*)d_in[6];
  enum { P_B1 = 0, P_B2, P_BOUT, P_LN1B, P_LN1G, P_LN2B, P_LN2G, P_W1, P_W2, P_WOUT, P_WQKV };
  auto LP = [&](int st, int layer, int which) -> const float* {
    return (const float*)d_in[7 + (st * 6 + layer) * 11 + which];
  };

  float* W = (float*)d_ws;
  float* out = (float*)d_out;

  // workspace layout (floats); big stage-2 temps alias stage-1 regions
  const long long RM = 9360;                       // max flat rows (24*390)
  const long long F_X0 = 0,            SZ_X = 8LL * 1025 * 256;
  const long long F_X1 = F_X0 + SZ_X;
  const long long F_XN = F_X1 + SZ_X,  SZ_XN = RM * 256;
  const long long F_QKV = F_XN + SZ_XN, SZ_QKV = RM * 768;
  const long long F_DOTS = F_QKV + SZ_QKV, SZ_DOTS = 8LL * 8 * 1025 * 1025;
  const long long F_AO = F_DOTS + SZ_DOTS, SZ_AO = RM * 256;
  const long long F_FFH = F_AO + SZ_AO, SZ_FFH = RM * 1024;
  const long long F_POS0 = F_FFH + SZ_FFH, SZ_POS = 8LL * 1024 * 256;
  const long long F_POS1 = F_POS0 + SZ_POS;
  const long long F_SCORE = F_POS1 + SZ_POS, SZ_SC = 8LL * 3 * 1024;
  const long long F_IDX = F_SCORE + SZ_SC;
  const long long F_KB = F_IDX + SZ_SC, SZ_KB = 8LL * 3 * 402 * 256;
  const long long F_VB = F_KB + SZ_KB;
  const long long F_CM = F_VB + SZ_KB;
  const long long F_GH = F_CM + 8LL * 512 * 768;
  const long long F_IMG = F_GH + 8LL * 512 * 128, SZ_IMG = 8LL * 3 * 512 * 256;
  const long long F_PG = F_IMG + SZ_IMG;
  const long long F_P2A = F_PG + SZ_IMG, SZ_P2 = 8LL * 3 * 384 * 256;
  const long long F_P2B = F_P2A + SZ_P2;
  const long long F_X2A = F_P2B + SZ_P2, SZ_X2 = 8LL * 3 * 390 * 256;
  const long long F_X2B = F_X2A + SZ_X2;

  float* xn = W + F_XN;  float* qkv = W + F_QKV;  float* dots = W + F_DOTS;
  float* ao = W + F_AO;  float* ffh = W + F_FFH;
  float* scoreb = W + F_SCORE;  int* idxb = (int*)(W + F_IDX);
  float* Kb = W + F_KB;  float* Vb = W + F_VB;
  float* cm = W + F_CM;  float* gh = W + F_GH;
  float* img = W + F_IMG; float* pg = W + F_PG;

  auto gemm = [&](const float* A, int lda, long long sA,
                  const float* Bm, int ldb, long long sB,
                  const float* bias, const float* Rr, long long sR,
                  float* Cc, int ldc, long long sC,
                  int M, int N, int Kd, int batch, int flags, float scale) {
    dim3 grid((unsigned)((N + 63) / 64), (unsigned)((M + 63) / 64), (unsigned)batch);
    gemm_wmma_bf16<<<grid, 128, 0, stream>>>(A, Bm, bias, Rr, Cc, M, N, Kd,
                                             lda, ldb, ldc, sA, sB, sC, sR, flags, scale);
  };

  const float SCALE = 0.0625f;  // 256^-0.5

  // d_out offsets: kp_body then attn slices per stage-1 layer
  long long aoff[6];
  aoff[0] = 8LL * 17 * 256;
  aoff[1] = aoff[0] + 8LL * 8 * 1 * 1024;
  aoff[2] = aoff[1] + 8LL * 8 * 1 * 1024;
  aoff[3] = aoff[2] + 8LL * 8 * 1 * 1024;
  aoff[4] = aoff[3] + 8LL * 8 * 1 * 768;
  aoff[5] = aoff[4] + 8LL * 8 * 17 * 768;

  // ---- init: x -> ws, pos broadcast over batch ----
  float* xcur = W + F_X0; float* xalt = W + F_X1;
  float* poscur = W + F_POS0; float* posalt = W + F_POS1;
  hipMemcpyAsync(xcur, x_in, SZ_X * sizeof(float), hipMemcpyDeviceToDevice, stream);
  copy_rows_k<<<dim3(1024, 8), 256, 0, stream>>>(pos_in, poscur, 256, 256, 256,
                                                 0LL, 1024LL * 256);

  // ============================ Stage 1 ============================
  int n = 1025, Kt = 1, nimg = 1024;
  const int tgt1[6] = {1024, 1024, 768, 768, 512, 512};

  for (int idx = 0; idx < 6; ++idx) {
    const float* wqkv = LP(0, idx, P_WQKV); const float* wout = LP(0, idx, P_WOUT);
    const float* bout = LP(0, idx, P_BOUT);
    const float* w1 = LP(0, idx, P_W1); const float* b1 = LP(0, idx, P_B1);
    const float* w2 = LP(0, idx, P_W2); const float* b2 = LP(0, idx, P_B2);

    if (idx > 0)
      add_rows_k<<<dim3(nimg, 8), 256, 0, stream>>>(xcur + (long long)Kt * 256, poscur, 256,
          256, 256, (long long)n * 256, (long long)nimg * 256);

    layernorm_k<<<8 * n, 256, 0, stream>>>(xcur, LP(0, idx, P_LN1G), LP(0, idx, P_LN1B), xn, 256);
    gemm(xn, 256, 0, wqkv, 768, 0, nullptr, nullptr, 0, qkv, 768, 0,
         8 * n, 768, 256, 1, 0, 0.0f);

    for (int h = 0; h < 8; ++h)   // dots = scale * Q K^T   (per head, batch=B)
      gemm(qkv + h * 32, 768, (long long)n * 768,
           qkv + 256 + h * 32, 768, (long long)n * 768,
           nullptr, nullptr, 0,
           dots + (long long)h * n * n, n, 8LL * n * n,
           n, n, 32, 8, FB_TRB | FB_SCL, SCALE);

    attn_slice_k<<<8 * 8 * Kt, 256, 0, stream>>>(dots, out + aoff[idx], n, Kt, n - Kt);

    bool prune = (nimg != tgt1[idx]);
    if (prune) {
      dim3 gs((nimg + 255) / 256, 8);
      score1_k<<<gs, 256, 0, stream>>>(dots, scoreb, n, Kt, nimg);
    }

    softmax_rows_k<<<8 * 8 * n, 256, 0, stream>>>(dots, n);

    for (int h = 0; h < 8; ++h)   // out = P V
      gemm(dots + (long long)h * n * n, n, 8LL * n * n,
           qkv + 512 + h * 32, 768, (long long)n * 768,
           nullptr, nullptr, 0,
           ao + h * 32, 256, (long long)n * 256,
           n, 32, n, 8, 0, 0.0f);

    gemm(ao, 256, 0, wout, 256, 0, bout, xcur, 0, xalt, 256, 0,
         8 * n, 256, 256, 1, FB_BIAS | FB_RES, 0.0f);
    swapf(xcur, xalt);

    if (prune) {
      int tg = tgt1[idx];
      topk_k<<<8, 256, 0, stream>>>(scoreb, idxb, nimg, tg);
      copy_rows_k<<<dim3(Kt, 8), 256, 0, stream>>>(xcur, xalt, 256, 256, 256,
          (long long)n * 256, (long long)(Kt + tg) * 256);
      gather_rows_k<<<dim3(tg, 8), 256, 0, stream>>>(xcur + (long long)Kt * 256, idxb,
          xalt + (long long)Kt * 256, 256,
          (long long)n * 256, (long long)(Kt + tg) * 256, tg);
      gather_rows_k<<<dim3(tg, 8), 256, 0, stream>>>(poscur, idxb, posalt, 256,
          (long long)nimg * 256, (long long)tg * 256, tg);
      swapf(poscur, posalt); swapf(xcur, xalt);
      nimg = tg; n = Kt + nimg;
    }

    if (idx == 3) {  // insert 17 sparse-kp tokens
      kp_insert_k<<<dim3(17, 8), 256, 0, stream>>>(xcur, skp, xalt,
          (long long)n * 256, (long long)(17 + nimg) * 256);
      copy_rows_k<<<dim3(nimg, 8), 256, 0, stream>>>(xcur + 256, xalt + 17LL * 256, 256,
          256, 256, (long long)n * 256, (long long)(17 + nimg) * 256);
      swapf(xcur, xalt);
      Kt = 17; n = 17 + nimg;
    }

    layernorm_k<<<8 * n, 256, 0, stream>>>(xcur, LP(0, idx, P_LN2G), LP(0, idx, P_LN2B), xn, 256);
    gemm(xn, 256, 0, w1, 1024, 0, b1, nullptr, 0, ffh, 1024, 0,
         8 * n, 1024, 256, 1, FB_BIAS | FB_GELU, 0.0f);
    gemm(ffh, 1024, 0, w2, 256, 0, b2, xcur, 0, xalt, 256, 0,
         8 * n, 256, 1024, 1, FB_BIAS | FB_RES, 0.0f);
    swapf(xcur, xalt);
  }

  // ===================== Stage-1 -> Stage-2 bridge =====================
  // n = 529, Kt = 17, nimg = 512 here
  int n2 = 390;
  float* x2cur = W + F_X2A; float* x2alt = W + F_X2B;
  float* p2cur = W + F_P2A; float* p2alt = W + F_P2B;

  kp_to_x2_k<<<dim3(18, 8), 256, 0, stream>>>(xcur, x2cur, (long long)n * 256, n2);

  // cm = sigmoid(relu(img @ w1 + b1) @ w2 + b2)
  gemm(xcur + 17LL * 256, 256, (long long)n * 256, gm_w1, 128, 0,
       gm_b1, nullptr, 0, gh, 128, 512LL * 128,
       512, 128, 256, 8, FB_BIAS | FB_RELU, 0.0f);
  gemm(gh, 128, 512LL * 128, gm_w2, 768, 0,
       gm_b2, nullptr, 0, cm, 768, 512LL * 768,
       512, 768, 128, 8, FB_BIAS | FB_SIG, 0.0f);

  mul_gate_k<<<dim3(512, 24), 256, 0, stream>>>(xcur + 17LL * 256, poscur, cm, img, pg,
                                                512, (long long)n * 256);

  score_pre2_k<<<dim3((512 + 255) / 256, 24), 256, 0, stream>>>(out + aoff[5], scoreb, 512);
  topk_k<<<24, 256, 0, stream>>>(scoreb, idxb, 512, 384);
  gather_rows_k<<<dim3(384, 24), 256, 0, stream>>>(img, idxb, x2cur + 6LL * 256, 256,
      512LL * 256, (long long)n2 * 256, 384);
  gather_rows_k<<<dim3(384, 24), 256, 0, stream>>>(pg, idxb, p2cur, 256,
      512LL * 256, 384LL * 256, 384);

  // ============================ Stage 2 ============================
  float* xn2 = W + F_XN;  float* qkv2 = W + F_QKV;   // alias stage-1 temps
  float* d2 = W + F_DOTS; float* ao2 = W + F_AO; float* ffh2 = W + F_FFH;
  int now2 = 384;
  const int tgt2[6] = {384, 384, 256, 256, 192, 192};

  for (int idx = 0; idx < 6; ++idx) {
    const float* wqkv = LP(1, idx, P_WQKV); const float* wout = LP(1, idx, P_WOUT);
    const float* bout = LP(1, idx, P_BOUT);
    const float* w1 = LP(1, idx, P_W1); const float* b1 = LP(1, idx, P_B1);
    const float* w2 = LP(1, idx, P_W2); const float* b2 = LP(1, idx, P_B2);
    int nk2 = n2 + 12;

    add_rows_k<<<dim3(now2, 24), 256, 0, stream>>>(x2cur + 6LL * 256, p2cur, 256,
        256, 256, (long long)n2 * 256, (long long)now2 * 256);

    layernorm_k<<<24 * n2, 256, 0, stream>>>(x2cur, LP(1, idx, P_LN1G), LP(1, idx, P_LN1B), xn2, 256);
    gemm(xn2, 256, 0, wqkv, 768, 0, nullptr, nullptr, 0, qkv2, 768, 0,
         24 * n2, 768, 256, 1, 0, 0.0f);

    build_kv_k<<<dim3(nk2, 24), 256, 0, stream>>>(qkv2, Kb, Vb, n2, nk2);

    for (int h = 0; h < 8; ++h)
      gemm(qkv2 + h * 32, 768, (long long)n2 * 768,
           Kb + h * 32, 256, (long long)nk2 * 256,
           nullptr, nullptr, 0,
           d2 + (long long)h * n2 * nk2, nk2, 8LL * n2 * nk2,
           n2, nk2, 32, 24, FB_TRB | FB_SCL, SCALE);

    long long tot = 24LL * 8 * n2 * nk2;
    group_mask_k<<<(unsigned)((tot + 255) / 256), 256, 0, stream>>>(d2, n2, nk2, tot);

    bool prune = (now2 != tgt2[idx]);
    if (prune)
      score_group2_k<<<dim3((now2 + 255) / 256, 24), 256, 0, stream>>>(d2, scoreb, n2, nk2);

    softmax_rows_k<<<24 * 8 * n2, 256, 0, stream>>>(d2, nk2);

    for (int h = 0; h < 8; ++h)
      gemm(d2 + (long long)h * n2 * nk2, nk2, 8LL * n2 * nk2,
           Vb + h * 32, 256, (long long)nk2 * 256,
           nullptr, nullptr, 0,
           ao2 + h * 32, 256, (long long)n2 * 256,
           n2, 32, nk2, 24, 0, 0.0f);

    gemm(ao2, 256, 0, wout, 256, 0, bout, x2cur, 0, x2alt, 256, 0,
         24 * n2, 256, 256, 1, FB_BIAS | FB_RES, 0.0f);
    swapf(x2cur, x2alt);

    if (prune) {
      int tg = tgt2[idx];
      topk_k<<<24, 256, 0, stream>>>(scoreb, idxb, now2, tg);
      copy_rows_k<<<dim3(6, 24), 256, 0, stream>>>(x2cur, x2alt, 256, 256, 256,
          (long long)n2 * 256, (long long)(6 + tg) * 256);
      gather_rows_k<<<dim3(tg, 24), 256, 0, stream>>>(x2cur + 6LL * 256, idxb,
          x2alt + 6LL * 256, 256, (long long)n2 * 256, (long long)(6 + tg) * 256, tg);
      gather_rows_k<<<dim3(tg, 24), 256, 0, stream>>>(p2cur, idxb, p2alt, 256,
          (long long)now2 * 256, (long long)tg * 256, tg);
      swapf(x2cur, x2alt); swapf(p2cur, p2alt);
      now2 = tg; n2 = 6 + tg;
    }

    layernorm_k<<<24 * n2, 256, 0, stream>>>(x2cur, LP(1, idx, P_LN2G), LP(1, idx, P_LN2B), xn2, 256);
    gemm(xn2, 256, 0, w1, 1024, 0, b1, nullptr, 0, ffh2, 1024, 0,
         24 * n2, 1024, 256, 1, FB_BIAS | FB_GELU, 0.0f);
    gemm(ffh2, 1024, 0, w2, 256, 0, b2, x2cur, 0, x2alt, 256, 0,
         24 * n2, 256, 1024, 1, FB_BIAS | FB_RES, 0.0f);
    swapf(x2cur, x2alt);
  }

  kp_out_k<<<dim3(17, 8), 256, 0, stream>>>(x2cur, out, n2);
}
